// R_GAT_87720412054006
// MI455X (gfx1250) — matrix operand order repaired and checked by reading, not compiled, not run on hardware
//
#include <hip/hip_runtime.h>
#include <cstdint>
#include <cstddef>

// ---------------------------------------------------------------------------
// R-GAT (2-layer relational GATConv + linear head) for MI455X (gfx1250).
// Dense transforms: v_wmma_f32_16x16x32_bf16, weights pre-packed into
// fragment-major layout so the GEMM inner loop is pure b128 loads + WMMA;
// K, NC, BIAS are compile-time so the epilogue is branch-free stores.
// Edge softmax/aggregation: L2-resident gathers + f32/u32 global atomics.
// ---------------------------------------------------------------------------

static constexpr int kR     = 3;
static constexpr int kN     = 50000;   // divisible by 16 (3125 M-tiles)
static constexpr int kE     = 800000;
static constexpr int kIN    = 128;
static constexpr int kHID   = 64;
static constexpr int kHEADS = 2;
static constexpr int kOUT   = 64;

static_assert(kN % 16 == 0, "M tiles must cover kN exactly");

typedef __attribute__((ext_vector_type(16))) __bf16       v16bf;
typedef __attribute__((ext_vector_type(8)))  float        v8f;
typedef __attribute__((ext_vector_type(4)))  unsigned int u32x4;
typedef __attribute__((ext_vector_type(4)))  float        f32x4;

union Frag { v16bf v; u32x4 u[2]; };

// ------------------------------ small utils -------------------------------

__global__ void fill_u32_kernel(uint32_t* p, uint32_t v, size_t n) {
  size_t i = (size_t)blockIdx.x * blockDim.x + threadIdx.x;
  if (i < n) p[i] = v;
}

__global__ void cast_bf16_kernel(const float* __restrict__ in, __bf16* __restrict__ out, size_t n) {
  size_t i = (size_t)blockIdx.x * blockDim.x + threadIdx.x;
  if (i < n) out[i] = (__bf16)in[i];
}

// --------------------- weight pack: fp32 -> WMMA B-frag --------------------
// Packed layout: fragment f = ntile*(K/32) + kb ; per-lane 16 bf16 chunk at
// element offset ((f*32 + lane)*16). Fragment element j of lane:
//   n = ntile*16 + (lane&15),  k = kb*32 + (j>>3)*16 + (lane>>4)*8 + (j&7)   (B is [K,NC] RM)
__global__ void pack_w_kernel(const float* __restrict__ W, __bf16* __restrict__ Bp,
                              int K, int NC) {
  int o = blockIdx.x * blockDim.x + threadIdx.x;
  if (o >= K * NC) return;
  int j    = o & 15;
  int lane = (o >> 4) & 31;
  int f    = o >> 9;
  int KB   = K / 32;
  int kb   = f % KB;
  int t    = f / KB;
  int n    = t * 16 + (lane & 15);
  int k    = kb * 32 + (j >> 3) * 16 + (lane >> 4) * 8 + (j & 7);
  Bp[o] = (__bf16)W[k * NC + n];
}

// ------------------------------ WMMA GEMM ---------------------------------
// C[M,NC] = A[M,K](bf16 RM) @ B(packed frags) (+ bias[NC]).
// One wave = one 16x16 tile; K fully unrolled (K/32 WMMAs).
// A frag (ISA 7.12.2): lane = 16*half + (M%16); elems 0..7 <-> K=half*8+j,
// elems 8..15 <-> K=16+half*8+(j-8)  -> two 16B contiguous loads per k-step.
// Requires: M % 16 == 0, grid guarantees ntile*16 < NC, EXEC all-ones.
template <int K, int NC, bool BIAS>
__global__ void wmma_gemm_bf16_kernel(const __bf16* __restrict__ A,
                                      const __bf16* __restrict__ Bp,
                                      const float* __restrict__ bias,
                                      float* __restrict__ C) {
  constexpr int KB = K / 32;
  const int lane  = threadIdx.x & 31;
  const int wave  = threadIdx.x >> 5;
  const int ntile = blockIdx.x * 4 + wave;
  const int m0    = blockIdx.y * 16;
  const int half  = lane >> 4;
  const int l15   = lane & 15;
  const int arow  = m0 + l15;
  const int col   = ntile * 16 + l15;

  const __bf16* arowp = A + (size_t)arow * K;
  const u32x4*  bfrag = (const u32x4*)Bp + ((size_t)ntile * KB * 32 + lane) * 2;

  v8f acc = {};
#pragma unroll
  for (int kb = 0; kb < KB; ++kb) {
    Frag a, b;
    const __bf16* ap = arowp + kb * 32 + half * 8;   // 16B aligned
    a.u[0] = *(const u32x4*)(ap);
    a.u[1] = *(const u32x4*)(ap + 16);
    b.u[0] = bfrag[(size_t)kb * 64 + 0];
    b.u[1] = bfrag[(size_t)kb * 64 + 1];
    acc = __builtin_amdgcn_wmma_f32_16x16x32_bf16(
        /*neg_a=*/false, a.v, /*neg_b=*/false, b.v,
        /*c_mod=*/(short)0, acc, /*reuse_a=*/false, /*reuse_b=*/false);
  }

  float badd = 0.0f;
  if constexpr (BIAS) badd = bias[col];
  float* crow = C + (size_t)(m0 + half * 8) * NC + col;   // 8 rows, stride NC
#pragma unroll
  for (int v = 0; v < 8; ++v) crow[(size_t)v * NC] = acc[v] + badd;
}

// --------------------------- attention scores -----------------------------

__global__ void attn_score_kernel(const float* __restrict__ feat,
                                  const float* __restrict__ al,
                                  const float* __restrict__ ar,
                                  float* __restrict__ el, float* __restrict__ er,
                                  int n, int heads, int dim) {
  int i = blockIdx.x * blockDim.x + threadIdx.x;
  if (i >= n * heads) return;
  int node = i / heads, h = i % heads;
  const float* f = feat + (size_t)node * heads * dim + (size_t)h * dim;
  float sl = 0.f, sr = 0.f;
  for (int d = 0; d < dim; ++d) {
    float v = f[d];
    sl += v * al[h * dim + d];
    sr += v * ar[h * dim + d];
  }
  el[i] = sl; er[i] = sr;
}

// ------------------------- edge softmax (segment) -------------------------

__device__ __forceinline__ float lrelu(float x) { return x > 0.f ? x : 0.2f * x; }
// order-preserving float -> uint map for atomicMax
__device__ __forceinline__ uint32_t fmap(float f) {
  uint32_t u = __float_as_uint(f);
  return (u & 0x80000000u) ? ~u : (u | 0x80000000u);
}
__device__ __forceinline__ float funmap(uint32_t u) {
  uint32_t b = (u & 0x80000000u) ? (u ^ 0x80000000u) : ~u;
  return __uint_as_float(b);
}
#define NEG_INF_MAPPED 0x007FFFFFu   // fmap(-inf)

__global__ void edge_max_kernel(const int* __restrict__ src, const int* __restrict__ dst,
                                const float* __restrict__ el, const float* __restrict__ er,
                                uint32_t* __restrict__ mbits, int E, int heads) {
  int i = blockIdx.x * blockDim.x + threadIdx.x;
  if (i >= E * heads) return;
  int e = i / heads, h = i % heads;
  int s = src[e], d = dst[e];
  float v = lrelu(el[s * heads + h] + er[d * heads + h]);
  atomicMax(&mbits[d * heads + h], fmap(v));
}

__global__ void edge_expsum_kernel(const int* __restrict__ src, const int* __restrict__ dst,
                                   const float* __restrict__ el, const float* __restrict__ er,
                                   const uint32_t* __restrict__ mbits,
                                   float* __restrict__ ee, float* __restrict__ z,
                                   int E, int heads) {
  int i = blockIdx.x * blockDim.x + threadIdx.x;
  if (i >= E * heads) return;
  int e = i / heads, h = i % heads;
  int s = src[e], d = dst[e];
  float v = lrelu(el[s * heads + h] + er[d * heads + h]);
  float m = funmap(mbits[d * heads + h]);
  float x = __expf(v - m);
  ee[i] = x;
  atomicAdd(&z[d * heads + h], x);
}

// 16 lanes (H=1) / 32 lanes (H=2) cooperate per edge: float4 gather + scatter.
__global__ void edge_scatter_kernel(const int* __restrict__ src, const int* __restrict__ dst,
                                    const float* __restrict__ ee, const float* __restrict__ z,
                                    const float* __restrict__ feat, float* __restrict__ agg,
                                    int E, int heads, int dim) {
  const int per = heads * dim / 4;                 // float4 chunks per edge
  long long total = (long long)E * per;
  long long i = (long long)blockIdx.x * blockDim.x + threadIdx.x;
  if (i >= total) return;
  int e = (int)(i / per), r = (int)(i % per);
  int h = r / (dim / 4);
  int s = src[e], d = dst[e];
  float a = ee[(size_t)e * heads + h] / z[(size_t)d * heads + h];
  f32x4 f = reinterpret_cast<const f32x4*>(feat)[(size_t)s * per + r];
  float* o = agg + (size_t)d * heads * dim + (size_t)r * 4;
  atomicAdd(o + 0, f.x * a);
  atomicAdd(o + 1, f.y * a);
  atomicAdd(o + 2, f.z * a);
  atomicAdd(o + 3, f.w * a);
}

// out[i] += elu(agg[i] + bias[i & mask])   (per-relation bias+elu, then sum)
__global__ void elu_accum_kernel(const float* __restrict__ agg, const float* __restrict__ bias,
                                 float* __restrict__ out, size_t n, int hdmask) {
  size_t i = (size_t)blockIdx.x * blockDim.x + threadIdx.x;
  if (i >= n) return;
  float x = agg[i] + bias[(int)(i & (size_t)hdmask)];
  out[i] += (x > 0.f) ? x : (__expf(x) - 1.0f);
}

// ------------------------------- launcher ---------------------------------

extern "C" void kernel_launch(void* const* d_in, const int* in_sizes, int n_in,
                              void* d_out, int out_size, void* d_ws, size_t ws_size,
                              hipStream_t stream) {
  (void)in_sizes; (void)n_in; (void)out_size; (void)ws_size;
  const float* x   = (const float*)d_in[0];
  const int*   src = (const int*)  d_in[1];
  const int*   dst = (const int*)  d_in[2];
  const float* W1  = (const float*)d_in[3];
  const float* al1 = (const float*)d_in[4];
  const float* ar1 = (const float*)d_in[5];
  const float* b1  = (const float*)d_in[6];
  const float* W2  = (const float*)d_in[7];
  const float* al2 = (const float*)d_in[8];
  const float* ar2 = (const float*)d_in[9];
  const float* b2  = (const float*)d_in[10];
  const float* Wl  = (const float*)d_in[11];
  const float* bl  = (const float*)d_in[12];
  float* out = (float*)d_out;

  char* ws = (char*)d_ws;
  size_t off = 0;
  auto take = [&](size_t bytes) -> char* {
    char* p = ws + off;
    off = (off + bytes + 255) & ~(size_t)255;
    return p;
  };

  __bf16*   xb   = (__bf16*)  take((size_t)kN * kIN * 2);
  __bf16*   w1p  = (__bf16*)  take((size_t)kR * kIN * kHEADS * kHID * 2);  // packed frags
  __bf16*   w2p  = (__bf16*)  take((size_t)kR * kHEADS * kHID * kHID * 2); // packed frags
  __bf16*   wlp  = (__bf16*)  take((size_t)kHID * kOUT * 2);               // packed frags
  __bf16*   h1b  = (__bf16*)  take((size_t)kN * kHEADS * kHID * 2);
  __bf16*   h2b  = (__bf16*)  take((size_t)kN * kHID * 2);
  float*    feat = (float*)   take((size_t)kN * kIN * 4);
  float*    el   = (float*)   take((size_t)kN * kHEADS * 4);
  float*    er   = (float*)   take((size_t)kN * kHEADS * 4);
  uint32_t* mb   = (uint32_t*)take((size_t)kN * kHEADS * 4);
  float*    z    = (float*)   take((size_t)kN * kHEADS * 4);
  float*    ee   = (float*)   take((size_t)kE * kHEADS * 4);
  float*    aggR = (float*)   take((size_t)kN * kHEADS * kHID * 4);
  float*    h1   = (float*)   take((size_t)kN * kHEADS * kHID * 4);
  float*    h2   = (float*)   take((size_t)kN * kHID * 4);

  const int BS = 256;
  auto blk = [&](long long n) { return (unsigned)((n + BS - 1) / BS); };

  // A-side bf16 copy + weight fragment packing
  cast_bf16_kernel<<<blk((long long)kN * kIN), BS, 0, stream>>>(x, xb, (size_t)kN * kIN);
  for (int r = 0; r < kR; ++r) {
    pack_w_kernel<<<blk(kIN * kHEADS * kHID), BS, 0, stream>>>(
        W1 + (size_t)r * kIN * kHEADS * kHID, w1p + (size_t)r * kIN * kHEADS * kHID,
        kIN, kHEADS * kHID);
    pack_w_kernel<<<blk(kHEADS * kHID * kHID), BS, 0, stream>>>(
        W2 + (size_t)r * kHEADS * kHID * kHID, w2p + (size_t)r * kHEADS * kHID * kHID,
        kHEADS * kHID, kHID);
  }
  pack_w_kernel<<<blk(kHID * kOUT), BS, 0, stream>>>(Wl, wlp, kHID, kOUT);

  // h1 = 0
  fill_u32_kernel<<<blk((long long)kN * kHEADS * kHID), BS, 0, stream>>>(
      (uint32_t*)h1, 0u, (size_t)kN * kHEADS * kHID);

  const dim3 gemm128(2, kN / 16);   // NC=128: 8 ntiles / 4 waves
  const dim3 gemm64(1, kN / 16);    // NC=64 : 4 ntiles / 4 waves

  // ------------------------------ layer 1 ------------------------------
  for (int r = 0; r < kR; ++r) {
    const int H = kHEADS, D = kHID, F = H * D;  // F = 128
    wmma_gemm_bf16_kernel<kIN, kHEADS * kHID, false><<<gemm128, 128, 0, stream>>>(
        xb, w1p + (size_t)r * kIN * F, nullptr, feat);
    attn_score_kernel<<<blk((long long)kN * H), BS, 0, stream>>>(
        feat, al1 + (size_t)r * H * D, ar1 + (size_t)r * H * D, el, er, kN, H, D);
    fill_u32_kernel<<<blk((long long)kN * H), BS, 0, stream>>>(mb, NEG_INF_MAPPED, (size_t)kN * H);
    fill_u32_kernel<<<blk((long long)kN * H), BS, 0, stream>>>((uint32_t*)z, 0u, (size_t)kN * H);
    fill_u32_kernel<<<blk((long long)kN * F), BS, 0, stream>>>((uint32_t*)aggR, 0u, (size_t)kN * F);
    const int* sr = src + (size_t)r * kE;
    const int* dr = dst + (size_t)r * kE;
    edge_max_kernel<<<blk((long long)kE * H), BS, 0, stream>>>(sr, dr, el, er, mb, kE, H);
    edge_expsum_kernel<<<blk((long long)kE * H), BS, 0, stream>>>(sr, dr, el, er, mb, ee, z, kE, H);
    edge_scatter_kernel<<<blk((long long)kE * (F / 4)), BS, 0, stream>>>(sr, dr, ee, z, feat, aggR, kE, H, D);
    elu_accum_kernel<<<blk((long long)kN * F), BS, 0, stream>>>(aggR, b1 + (size_t)r * F, h1, (size_t)kN * F, F - 1);
  }
  cast_bf16_kernel<<<blk((long long)kN * kIN), BS, 0, stream>>>(h1, h1b, (size_t)kN * kIN);

  // h2 = 0
  fill_u32_kernel<<<blk((long long)kN * kHID), BS, 0, stream>>>((uint32_t*)h2, 0u, (size_t)kN * kHID);

  // ------------------------------ layer 2 ------------------------------
  for (int r = 0; r < kR; ++r) {
    const int H = 1, D = kHID, F = D;  // F = 64
    wmma_gemm_bf16_kernel<kHEADS * kHID, kHID, false><<<gemm64, 128, 0, stream>>>(
        h1b, w2p + (size_t)r * (kHEADS * kHID) * D, nullptr, feat);
    attn_score_kernel<<<blk((long long)kN * H), BS, 0, stream>>>(
        feat, al2 + (size_t)r * D, ar2 + (size_t)r * D, el, er, kN, H, D);
    fill_u32_kernel<<<blk((long long)kN * H), BS, 0, stream>>>(mb, NEG_INF_MAPPED, (size_t)kN * H);
    fill_u32_kernel<<<blk((long long)kN * H), BS, 0, stream>>>((uint32_t*)z, 0u, (size_t)kN * H);
    fill_u32_kernel<<<blk((long long)kN * F), BS, 0, stream>>>((uint32_t*)aggR, 0u, (size_t)kN * F);
    const int* sr = src + (size_t)r * kE;
    const int* dr = dst + (size_t)r * kE;
    edge_max_kernel<<<blk((long long)kE * H), BS, 0, stream>>>(sr, dr, el, er, mb, kE, H);
    edge_expsum_kernel<<<blk((long long)kE * H), BS, 0, stream>>>(sr, dr, el, er, mb, ee, z, kE, H);
    edge_scatter_kernel<<<blk((long long)kE * (F / 4)), BS, 0, stream>>>(sr, dr, ee, z, feat, aggR, kE, H, D);
    elu_accum_kernel<<<blk((long long)kN * F), BS, 0, stream>>>(aggR, b2 + (size_t)r * F, h2, (size_t)kN * F, F - 1);
  }
  cast_bf16_kernel<<<blk((long long)kN * kHID), BS, 0, stream>>>(h2, h2b, (size_t)kN * kHID);

  // ------------------------- final linear head -------------------------
  {
    wmma_gemm_bf16_kernel<kHID, kOUT, true><<<gemm64, 128, 0, stream>>>(h2b, wlp, bl, out);
  }
}
